// ProteinAtomicEmbedder_37134287242038
// MI455X (gfx1250) — matrix operand order repaired and checked by reading, not compiled, not run on hardware
//
#include <hip/hip_runtime.h>
#include <math.h>

// ---------------------------------------------------------------------------
// ProteinAtomicEmbedder for MI455X (gfx1250, wave32, WMMA).
// All dense GEMMs use v_wmma_f32_16x16x32_f16 with:
//   - M, N forced to multiples of 16, K padded to multiples of 32 (zero tails)
//   - B pre-packed to f16 in the exact WMMA lane layout -> 2x b128 loads/tile
//   - A loaded as 4x b128 f32 + cvt (contiguous 8-float runs per lane)
// so the inner loop has NO bounds checks, NO exec masking, NO per-load waits.
// Per-edge TP weights are produced in <=27MB chunks that stay L2-resident and
// are consumed immediately by the tensor-product scatter kernel.
// ---------------------------------------------------------------------------

#define H_EDGE 16
#define ATOM_IN 16
#define BOND_IN 8
#define NUM_AA 21
#define N_LAYERS 5
#define NA 8192
#define NR 1024
#define EB 16384
#define ER 32768
#define ERES 16384
#define ATOM_RBF 16
#define RES_RBF 32
#define POS_EMB 32

typedef __attribute__((ext_vector_type(16))) _Float16 v16h;
typedef __attribute__((ext_vector_type(8)))  float    v8f;

// ===========================================================================
// Weight pre-packing: f32 [Ko x No] row-major -> f16 WMMA-B tile layout,
// zero-padded to [Kp x Np].  Packed element index:
//   i = ((kb*ntiles + nt)*32 + lane)*16 + e
//   k = kb*32 + (e/8)*16 + (lane/16)*8 + e%8 ;  n = nt*16 + lane%16
// ===========================================================================
__global__ __launch_bounds__(256) void k_pack_w(
    const float* __restrict__ W, _Float16* __restrict__ out,
    int Ko, int No, int Kp, int Np)
{
  int i = blockIdx.x * blockDim.x + threadIdx.x;
  if (i >= Kp * Np) return;
  int e    = i & 15;
  int lane = (i >> 4) & 31;
  int t    = i >> 9;
  int ntiles = Np >> 4;
  int kb = t / ntiles, nt = t % ntiles;
  int k = kb * 32 + ((e >> 3) << 4) + ((lane >> 4) << 3) + (e & 7);
  int n = (nt << 4) + (lane & 15);
  float v = (k < Ko && n < No) ? W[(size_t)k * No + n] : 0.0f;
  out[i] = (_Float16)v;
}

__global__ __launch_bounds__(256) void k_pack_b(
    const float* __restrict__ b, float* __restrict__ out, int No, int Np)
{
  int i = blockIdx.x * blockDim.x + threadIdx.x;
  if (i >= Np) return;
  out[i] = (i < No) ? b[i] : 0.0f;
}

// pad rows [M x No] -> [M x Np] with zero tails
__global__ __launch_bounds__(256) void k_pad_rows(
    const float* __restrict__ X, float* __restrict__ Y, int M, int No, int Np)
{
  int i = blockIdx.x * blockDim.x + threadIdx.x;
  if (i >= M * Np) return;
  int r = i / Np, c = i % Np;
  Y[i] = (c < No) ? X[(size_t)r * No + c] : 0.0f;
}

// ===========================================================================
// GEMM: Y[M x N] = act(X[M x K] @ Bpk + bias)
// Preconditions: M%16==0, N%16==0, K%32==0, X row stride = K (zero-padded),
// Bpk packed f16 (k_pack_w), bias length >= N.
// One wave per 16x16 output tile.
// ===========================================================================
__global__ __launch_bounds__(256) void k_gemm_wmma(
    const float* __restrict__ X, const _Float16* __restrict__ Bpk,
    const float* __restrict__ bias, float* __restrict__ Y,
    int M, int K, int N, int act)
{
  int wid  = (blockIdx.x * blockDim.x + threadIdx.x) >> 5;  // wave-uniform
  int lane = threadIdx.x & 31;
  int nt_n = N >> 4;
  if (wid >= (M >> 4) * nt_n) return;   // whole wave exits together
  int mt = wid / nt_n, nt = wid % nt_n;

  int hi  = lane >> 4;
  int row = (mt << 4) + (lane & 15);
  const float* xrow = X + (size_t)row * K + (hi << 3);
  const v16h*  bt   = reinterpret_cast<const v16h*>(Bpk) + (size_t)nt * 32 + lane;
  size_t bstep = (size_t)nt_n * 32;     // v16h units per k-block

  v8f acc = {};
  for (int kb = 0; kb < K; kb += 32) {
    float4 q0 = *(const float4*)(xrow + kb);
    float4 q1 = *(const float4*)(xrow + kb + 4);
    float4 q2 = *(const float4*)(xrow + kb + 16);
    float4 q3 = *(const float4*)(xrow + kb + 20);
    float r0[8], r1[8];
    *(float4*)&r0[0] = q0; *(float4*)&r0[4] = q1;
    *(float4*)&r1[0] = q2; *(float4*)&r1[4] = q3;
    v16h a;
#pragma unroll
    for (int e = 0; e < 8; ++e) {
      a[e]     = (_Float16)r0[e];
      a[8 + e] = (_Float16)r1[e];
    }
    v16h b = *bt;
    bt += bstep;
    acc = __builtin_amdgcn_wmma_f32_16x16x32_f16(false, a, false, b,
                                                 (short)0, acc, false, false);
  }
  // C/D layout: VGPR r -> m = mt*16 + r + 8*hi, n = nt*16 + lane%16
  int n = (nt << 4) + (lane & 15);
  float bn = bias[n];
#pragma unroll
  for (int r = 0; r < 8; ++r) {
    int m = (mt << 4) + r + (hi << 3);
    float v = acc[r] + bn;
    if (act) v = fmaxf(v, 0.0f);
    Y[(size_t)m * N + n] = v;
  }
}

// ===========================================================================
// LayerNorm over first N of each row (row stride ld), one thread per row.
// ===========================================================================
__global__ __launch_bounds__(256) void k_ln(
    float* __restrict__ Y, const float* __restrict__ g,
    const float* __restrict__ b, int M, int N, int ld)
{
  int r = blockIdx.x * blockDim.x + threadIdx.x;
  if (r >= M) return;
  float* y = Y + (size_t)r * ld;
  float mu = 0.f;
  for (int j = 0; j < N; ++j) mu += y[j];
  mu /= N;
  float var = 0.f;
  for (int j = 0; j < N; ++j) { float d = y[j] - mu; var += d * d; }
  var /= N;
  float inv = rsqrtf(var + 1e-5f);
  for (int j = 0; j < N; ++j) y[j] = (y[j] - mu) * inv * g[j] + b[j];
}

// ===========================================================================
// Feature kernels
// ===========================================================================
__device__ static inline void rbf_write(float d, float dmax, int n, float* out) {
  float sig = dmax / n;
  for (int j = 0; j < n; ++j) {
    float mu = dmax * (float)j / (float)(n - 1);
    float t = (d - mu) / sig;
    out[j] = expf(-t * t);
  }
}
__device__ static inline float edist(float vx, float vy, float vz) {
  float ax = vx + 1e-12f, ay = vy + 1e-12f, az = vz + 1e-12f;
  return sqrtf(ax * ax + ay * ay + az * az);
}
__device__ static inline void sh_write(float vx, float vy, float vz, float* sh) {
  float n = sqrtf(vx * vx + vy * vy + vz * vz + 1e-12f);
  const float S3 = 1.7320508075688772f;
  sh[0] = 1.f; sh[1] = S3 * vx / n; sh[2] = S3 * vy / n; sh[3] = S3 * vz / n;
}

// one-hot with padded row stride kp (cols >= ko zeroed)
__global__ __launch_bounds__(256) void k_onehot(
    const int* __restrict__ seq, float* __restrict__ out, int n, int ko, int kp)
{
  int i = blockIdx.x * blockDim.x + threadIdx.x;
  if (i >= n * kp) return;
  int r = i / kp, c = i % kp;
  out[i] = (c < ko && seq[r] == c) ? 1.0f : 0.0f;
}

// res edges: row stride 128 = [rbf32 | pe32 | oh(dst)21 | oh(src)21 | 0...]
__global__ __launch_bounds__(256) void k_res_edge_feat(
    const float* __restrict__ coords, const int* __restrict__ ei,
    const float* __restrict__ res_in, float* __restrict__ edge_in,
    float* __restrict__ sh_out, int E)
{
  int e = blockIdx.x * blockDim.x + threadIdx.x;
  if (e >= E) return;
  int d = ei[e], s = ei[E + e];
  float vx = coords[d * 3 + 0] - coords[s * 3 + 0];
  float vy = coords[d * 3 + 1] - coords[s * 3 + 1];
  float vz = coords[d * 3 + 2] - coords[s * 3 + 2];
  float dist = edist(vx, vy, vz);
  float* o = edge_in + (size_t)e * 128;
  rbf_write(dist, 22.0f, RES_RBF, o);
  float dd = (float)(d - s);
  const float c = -logf(10000.0f) / (float)POS_EMB;
  for (int j = 0; j < POS_EMB / 2; ++j) {
    float ang = dd * expf((float)(2 * j) * c);
    o[RES_RBF + j]               = cosf(ang);
    o[RES_RBF + POS_EMB / 2 + j] = sinf(ang);
  }
  for (int j = 0; j < NUM_AA; ++j) {
    o[64 + j]          = res_in[d * 32 + j];   // res_in row stride 32
    o[64 + NUM_AA + j] = res_in[s * 32 + j];
  }
  for (int j = 106; j < 128; ++j) o[j] = 0.0f;
  sh_write(vx, vy, vz, sh_out + (size_t)e * 4);
}

// bond edges: row stride 64 = [bond 8 | rbf16 | ap(dst)16 | ap(dst)16 | 0x8]
__global__ __launch_bounds__(256) void k_bond_feat(
    const float* __restrict__ coords, const int* __restrict__ ei,
    const float* __restrict__ ap, const float* __restrict__ bp,
    float* __restrict__ edge_in, float* __restrict__ sh_out, int E)
{
  int e = blockIdx.x * blockDim.x + threadIdx.x;
  if (e >= E) return;
  int d = ei[e], s = ei[E + e];
  float vx = coords[d * 3 + 0] - coords[s * 3 + 0];
  float vy = coords[d * 3 + 1] - coords[s * 3 + 1];
  float vz = coords[d * 3 + 2] - coords[s * 3 + 2];
  float* o = edge_in + (size_t)e * 64;
  for (int j = 0; j < BOND_IN; ++j) o[j] = bp[(size_t)e * BOND_IN + j];
  rbf_write(edist(vx, vy, vz), 5.0f, ATOM_RBF, o + BOND_IN);
  for (int j = 0; j < ATOM_IN; ++j) {
    float v = ap[(size_t)d * ATOM_IN + j];
    o[BOND_IN + ATOM_RBF + j] = v;
    o[BOND_IN + ATOM_RBF + ATOM_IN + j] = v;
  }
  for (int j = 56; j < 64; ++j) o[j] = 0.0f;
  sh_write(vx, vy, vz, sh_out + (size_t)e * 4);
}

// radius edges: row stride 64 = [rbf16 | ap(dst)16 | ap(dst)16 | 0x16]
__global__ __launch_bounds__(256) void k_rad_feat(
    const float* __restrict__ coords, const int* __restrict__ ei,
    const float* __restrict__ ap, float* __restrict__ edge_in,
    float* __restrict__ sh_out, int E)
{
  int e = blockIdx.x * blockDim.x + threadIdx.x;
  if (e >= E) return;
  int d = ei[e], s = ei[E + e];
  float vx = coords[d * 3 + 0] - coords[s * 3 + 0];
  float vy = coords[d * 3 + 1] - coords[s * 3 + 1];
  float vz = coords[d * 3 + 2] - coords[s * 3 + 2];
  float* o = edge_in + (size_t)e * 64;
  rbf_write(edist(vx, vy, vz), 5.0f, ATOM_RBF, o);
  for (int j = 0; j < ATOM_IN; ++j) {
    float v = ap[(size_t)d * ATOM_IN + j];
    o[ATOM_RBF + j] = v;
    o[ATOM_RBF + ATOM_IN + j] = v;
  }
  for (int j = 48; j < 64; ++j) o[j] = 0.0f;
  sh_write(vx, vy, vz, sh_out + (size_t)e * 4);
}

// agg "edges": row stride 32 = [rbf16(d,10) | atom_props 16]
__global__ __launch_bounds__(256) void k_agg_feat(
    const float* __restrict__ ac, const float* __restrict__ rc,
    const int* __restrict__ arb, const float* __restrict__ ap,
    float* __restrict__ out, float* __restrict__ sh_out, int n)
{
  int a = blockIdx.x * blockDim.x + threadIdx.x;
  if (a >= n) return;
  int r = arb[a];
  float vx = ac[a * 3 + 0] - rc[r * 3 + 0];
  float vy = ac[a * 3 + 1] - rc[r * 3 + 1];
  float vz = ac[a * 3 + 2] - rc[r * 3 + 2];
  float* o = out + (size_t)a * 32;
  rbf_write(edist(vx, vy, vz), 10.0f, ATOM_RBF, o);
  for (int j = 0; j < ATOM_IN; ++j) o[ATOM_RBF + j] = ap[(size_t)a * ATOM_IN + j];
  sh_write(vx, vy, vz, sh_out + (size_t)a * 4);
}

// ===========================================================================
// Tensor product scatter (e3nn specs), fp32 + atomic segment sums.
// ===========================================================================
struct TpMeta {
  int si, vi1, vi2, pi;
  int so, vo1, vo2, po;
  int off[10];
  float scl[10];
  float gn[4];
  int numel, din, dout;
};

__global__ __launch_bounds__(256) void k_tp_scatter(
    const float* __restrict__ node, const int* __restrict__ src, int srcBase,
    const int* __restrict__ dst, const float* __restrict__ sh,
    const float* __restrict__ w, float* __restrict__ seg_sum,
    float* __restrict__ seg_cnt, int nE, TpMeta tm)
{
  int e = blockIdx.x * blockDim.x + threadIdx.x;
  if (e >= nE) return;
  int sn = src ? src[e] : (srcBase + e);
  int dn = dst[e];
  const float* x  = node + (size_t)sn * tm.din;
  const float* we = w + (size_t)e * tm.numel;
  const float* she = sh + (size_t)e * 4;
  float sh0 = she[0], hx = she[1], hy = she[2], hz = she[3];
  const float* xs  = x;
  const float* xv1 = x + tm.si;
  const float* xv2 = x + tm.si + 3 * tm.vi1;
  const float* xp  = x + tm.si + 3 * tm.vi1 + 3 * tm.vi2;
  float* orow = seg_sum + (size_t)dn * tm.dout;

  for (int o = 0; o < tm.so; ++o) {
    float acc = 0.f;
    if (tm.scl[0] != 0.f) {
      const float* W = we + tm.off[0];
      float t = 0.f;
      for (int i = 0; i < tm.si; ++i) t += xs[i] * W[i * tm.so + o];
      acc += t * sh0 * tm.scl[0];
    }
    if (tm.scl[1] != 0.f) {
      const float* W = we + tm.off[1];
      float t = 0.f;
      for (int i = 0; i < tm.vi1; ++i) {
        const float* v = xv1 + 3 * i;
        t += (v[0] * hx + v[1] * hy + v[2] * hz) * W[i * tm.so + o];
      }
      acc += t * tm.scl[1];
    }
    atomicAdd(&orow[o], acc * tm.gn[0]);
  }
  for (int o = 0; o < tm.vo1; ++o) {
    float ax = 0.f, ay = 0.f, az = 0.f;
    if (tm.scl[2] != 0.f) {
      const float* W = we + tm.off[2];
      float t = 0.f;
      for (int i = 0; i < tm.si; ++i) t += xs[i] * W[i * tm.vo1 + o];
      t *= tm.scl[2]; ax += t * hx; ay += t * hy; az += t * hz;
    }
    if (tm.scl[3] != 0.f) {
      const float* W = we + tm.off[3];
      float tx = 0.f, ty = 0.f, tz = 0.f;
      for (int i = 0; i < tm.vi1; ++i) {
        float wv = W[i * tm.vo1 + o]; const float* v = xv1 + 3 * i;
        tx += v[0] * wv; ty += v[1] * wv; tz += v[2] * wv;
      }
      float s = sh0 * tm.scl[3]; ax += tx * s; ay += ty * s; az += tz * s;
    }
    if (tm.scl[4] != 0.f) {
      const float* W = we + tm.off[4];
      float tx = 0.f, ty = 0.f, tz = 0.f;
      for (int i = 0; i < tm.vi2; ++i) {
        const float* v = xv2 + 3 * i;
        float cx = v[1] * hz - v[2] * hy;
        float cy = v[2] * hx - v[0] * hz;
        float cz = v[0] * hy - v[1] * hx;
        float wv = W[i * tm.vo1 + o];
        tx += cx * wv; ty += cy * wv; tz += cz * wv;
      }
      ax += tx * tm.scl[4]; ay += ty * tm.scl[4]; az += tz * tm.scl[4];
    }
    int b = tm.so + 3 * o;
    atomicAdd(&orow[b + 0], ax * tm.gn[1]);
    atomicAdd(&orow[b + 1], ay * tm.gn[1]);
    atomicAdd(&orow[b + 2], az * tm.gn[1]);
  }
  for (int o = 0; o < tm.vo2; ++o) {
    float ax = 0.f, ay = 0.f, az = 0.f;
    if (tm.scl[5] != 0.f) {
      const float* W = we + tm.off[5];
      float tx = 0.f, ty = 0.f, tz = 0.f;
      for (int i = 0; i < tm.vi1; ++i) {
        const float* v = xv1 + 3 * i;
        float cx = v[1] * hz - v[2] * hy;
        float cy = v[2] * hx - v[0] * hz;
        float cz = v[0] * hy - v[1] * hx;
        float wv = W[i * tm.vo2 + o];
        tx += cx * wv; ty += cy * wv; tz += cz * wv;
      }
      ax += tx * tm.scl[5]; ay += ty * tm.scl[5]; az += tz * tm.scl[5];
    }
    if (tm.scl[6] != 0.f) {
      const float* W = we + tm.off[6];
      float tx = 0.f, ty = 0.f, tz = 0.f;
      for (int i = 0; i < tm.vi2; ++i) {
        float wv = W[i * tm.vo2 + o]; const float* v = xv2 + 3 * i;
        tx += v[0] * wv; ty += v[1] * wv; tz += v[2] * wv;
      }
      float s = sh0 * tm.scl[6]; ax += tx * s; ay += ty * s; az += tz * s;
    }
    if (tm.scl[7] != 0.f) {
      const float* W = we + tm.off[7];
      float t = 0.f;
      for (int i = 0; i < tm.pi; ++i) t += xp[i] * W[i * tm.vo2 + o];
      t *= tm.scl[7]; ax += t * hx; ay += t * hy; az += t * hz;
    }
    int b = tm.so + 3 * tm.vo1 + 3 * o;
    atomicAdd(&orow[b + 0], ax * tm.gn[2]);
    atomicAdd(&orow[b + 1], ay * tm.gn[2]);
    atomicAdd(&orow[b + 2], az * tm.gn[2]);
  }
  for (int o = 0; o < tm.po; ++o) {
    float acc = 0.f;
    if (tm.scl[8] != 0.f) {
      const float* W = we + tm.off[8];
      float t = 0.f;
      for (int i = 0; i < tm.vi2; ++i) {
        const float* v = xv2 + 3 * i;
        t += (v[0] * hx + v[1] * hy + v[2] * hz) * W[i * tm.po + o];
      }
      acc += t * tm.scl[8];
    }
    if (tm.scl[9] != 0.f) {
      const float* W = we + tm.off[9];
      float t = 0.f;
      for (int i = 0; i < tm.pi; ++i) t += xp[i] * W[i * tm.po + o];
      acc += t * sh0 * tm.scl[9];
    }
    atomicAdd(&orow[tm.so + 3 * tm.vo1 + 3 * tm.vo2 + o], acc * tm.gn[3]);
  }
  if (seg_cnt) atomicAdd(&seg_cnt[dn], 1.0f);
}

__global__ __launch_bounds__(256) void k_seg_finalize(
    const float* __restrict__ sum, const float* __restrict__ cnt,
    const float* __restrict__ node_in, float* __restrict__ out,
    int n, int dout, int din)
{
  int i = blockIdx.x * blockDim.x + threadIdx.x;
  if (i >= n * dout) return;
  int r = i / dout, c = i % dout;
  float v = sum[i] / fmaxf(cnt[r], 1.0f);
  if (c < din) v += node_in[(size_t)r * din + c];
  out[i] = v;
}

// ===========================================================================
// Host side
// ===========================================================================
static const int AMt[6][4] = {{16,0,0,0},{16,4,0,0},{16,4,4,0},
                              {16,4,4,16},{16,4,4,16},{16,4,4,16}};
static const int RMt[6][4] = {{32,0,0,0},{32,8,0,0},{32,8,8,0},
                              {32,8,8,32},{32,8,8,32},{32,8,8,32}};
static int irdim(const int m[4]) { return m[0] + 3*m[1] + 3*m[2] + m[3]; }

static void tp_fill(TpMeta& t, const int mi[4], const int mo[4]) {
  t.si = mi[0]; t.vi1 = mi[1]; t.vi2 = mi[2]; t.pi = mi[3];
  t.so = mo[0]; t.vo1 = mo[1]; t.vo2 = mo[2]; t.po = mo[3];
  int pa[10] = {mi[0],mi[1],mi[0],mi[1],mi[2],mi[1],mi[2],mi[3],mi[2],mi[3]};
  int pb[10] = {mo[0],mo[0],mo[1],mo[1],mo[1],mo[2],mo[2],mo[2],mo[3],mo[3]};
  int off = 0;
  for (int k = 0; k < 10; ++k) {
    t.off[k] = off;
    if (pa[k] * pb[k] > 0) { t.scl[k] = 1.0f / sqrtf((float)pa[k]); off += pa[k]*pb[k]; }
    else t.scl[k] = 0.f;
  }
  t.numel = off;
  int grp[10] = {0,0,1,1,1,2,2,2,3,3};
  int gc[4]   = {0,0,0,0};
  for (int k = 0; k < 10; ++k) if (pa[k]*pb[k] > 0) gc[grp[k]]++;
  for (int g = 0; g < 4; ++g) t.gn[g] = gc[g] ? 1.0f / sqrtf((float)gc[g]) : 0.f;
  t.din = irdim(mi); t.dout = irdim(mo);
}

struct MlpP { const float *w1,*b1,*w2,*b2,*g,*b; };
struct FcP  { const float *w1,*b1,*w2,*b2; };
// jax tree_flatten sorts dict keys: mlp_ln -> [b, b1, b2, g, w1, w2]
static MlpP mlp_at(void* const* din, int base) {
  MlpP p;
  p.b  = (const float*)din[base + 0]; p.b1 = (const float*)din[base + 1];
  p.b2 = (const float*)din[base + 2]; p.g  = (const float*)din[base + 3];
  p.w1 = (const float*)din[base + 4]; p.w2 = (const float*)din[base + 5];
  return p;
}
// fc -> [b1, b2, w1, w2]
static FcP fc_at(void* const* din, int base) {
  FcP p;
  p.b1 = (const float*)din[base + 0]; p.b2 = (const float*)din[base + 1];
  p.w1 = (const float*)din[base + 2]; p.w2 = (const float*)din[base + 3];
  return p;
}

// packed GEMM pair (fc / mlp front-end)
struct GFc { const _Float16 *w1, *w2; const float *b1, *b2;
             int K1, N1, K2, N2; };

static void gemm(const float* X, const _Float16* B, const float* bias, float* Y,
                 int M, int K, int N, int act, hipStream_t s)
{
  long waves = (long)(M >> 4) * (N >> 4);
  long grid  = (waves * 32 + 255) / 256;
  k_gemm_wmma<<<(int)grid, 256, 0, s>>>(X, B, bias, Y, M, K, N, act);
}
static inline int gr(int n) { return (n + 255) / 256; }

// ===========================================================================
extern "C" void kernel_launch(void* const* d_in, const int* in_sizes, int n_in,
                              void* d_out, int out_size, void* d_ws, size_t ws_size,
                              hipStream_t stream)
{
  (void)in_sizes; (void)n_in; (void)ws_size;
  const float* atom_coords = (const float*)d_in[0];
  const float* res_coords  = (const float*)d_in[1];
  const float* atom_props  = (const float*)d_in[2];
  const float* bond_props  = (const float*)d_in[3];
  const int*   seq         = (const int*)d_in[4];
  const int*   bei         = (const int*)d_in[5];
  const int*   rei         = (const int*)d_in[6];
  const int*   sei         = (const int*)d_in[7];
  const int*   arb         = (const int*)d_in[8];

  // params flattened alphabetically: agg_edge_embed, atom_embed,
  // bond_edge_embed, layers[0..4]{fc_agg,fc_bond,fc_rad,fc_res},
  // radius_edge_embed, res_edge_embed, res_embed
  int P = 9;
  MlpP p_agg_e  = mlp_at(d_in, P); P += 6;
  MlpP p_atom_e = mlp_at(d_in, P); P += 6;
  MlpP p_bond_e = mlp_at(d_in, P); P += 6;
  FcP Lagg[N_LAYERS], Lbond[N_LAYERS], Lrad[N_LAYERS], Lres[N_LAYERS];
  for (int i = 0; i < N_LAYERS; ++i) {
    Lagg[i]  = fc_at(d_in, P); P += 4;
    Lbond[i] = fc_at(d_in, P); P += 4;
    Lrad[i]  = fc_at(d_in, P); P += 4;
    Lres[i]  = fc_at(d_in, P); P += 4;
  }
  MlpP p_rad_e  = mlp_at(d_in, P); P += 6;
  MlpP p_rese_e = mlp_at(d_in, P); P += 6;
  MlpP p_res_e  = mlp_at(d_in, P); P += 6;

  // ---- workspace bump allocator ----
  char* base = (char*)d_ws;
  size_t off = 0;
  auto alloc = [&](size_t nfloats) -> float* {
    float* p = (float*)(base + off);
    off += ((nfloats * sizeof(float)) + 255) & ~(size_t)255;
    return p;
  };
  auto alloch = [&](size_t nhalf) -> _Float16* {
    _Float16* p = (_Float16*)(base + off);
    off += ((nhalf * 2) + 255) & ~(size_t)255;
    return p;
  };
  auto pack_w = [&](const float* W, int Ko, int No, int Kp, int Np) -> const _Float16* {
    _Float16* dst = alloch((size_t)Kp * Np);
    k_pack_w<<<gr(Kp * Np), 256, 0, stream>>>(W, dst, Ko, No, Kp, Np);
    return dst;
  };
  auto pack_bias = [&](const float* b, int No, int Np) -> const float* {
    float* dst = alloc(Np);
    k_pack_b<<<gr(Np), 256, 0, stream>>>(b, dst, No, Np);
    return dst;
  };
  auto mk_fc = [&](const FcP& f, int k1o, int n1o, int n2) -> GFc {
    GFc g;
    g.K1 = 32; g.N1 = 32; g.K2 = 32; g.N2 = n2;
    g.w1 = pack_w(f.w1, k1o, n1o, 32, 32);
    g.b1 = pack_bias(f.b1, n1o, 32);
    g.w2 = pack_w(f.w2, n1o, n2, 32, n2);
    g.b2 = pack_bias(f.b2, n2, n2);
    return g;
  };

  float* res_in      = alloc((size_t)NR * 32);        // one-hot, padded 21->32
  float* res_edge_in = alloc((size_t)ERES * 128);     // 106 -> 128
  float* res_sh      = alloc((size_t)ERES * 4);
  float* res_edge_f  = alloc((size_t)ERES * 32);
  float* agg_in      = alloc((size_t)NA * 32);
  float* agg_sh      = alloc((size_t)NA * 4);
  float* agg_edge_f  = alloc((size_t)NA * 32);        // 16 real + 16 zero
  float* rad_in      = alloc((size_t)ER * 64);        // 48 -> 64
  float* rad_edge_f  = alloc((size_t)ER * 32);
  float* bond_in     = alloc((size_t)EB * 64);        // 56 -> 64
  float* bond_edge_f = alloc((size_t)EB * 32);
  float* ap_pad      = alloc((size_t)NA * 32);        // atom_props 16 -> 32
  float* atom_sh     = alloc((size_t)(EB + ER) * 4);
  float* atom_fA     = alloc((size_t)NA * 56);
  float* atom_fB     = alloc((size_t)NA * 56);
  float* res_fA      = alloc((size_t)NR * 112);
  float* res_fB      = alloc((size_t)NR * 112);
  float* seg_sum_a   = alloc((size_t)NA * 56);
  float* seg_cnt_a   = alloc((size_t)NA);
  float* seg_sum_r   = alloc((size_t)NR * 112);
  float* seg_cnt_r   = alloc((size_t)NR);
  float* h_buf       = alloc((size_t)32768 * 32);
  const size_t WCAP  = (size_t)2048 * 3328;
  float* wbuf        = alloc(WCAP);

  // ---- pack all weights (once per launch; all tiny) ----
  TpMeta tAm[N_LAYERS], tGm[N_LAYERS], tRm[N_LAYERS];
  GFc gBond[N_LAYERS], gRad[N_LAYERS], gAgg[N_LAYERS], gRes[N_LAYERS];
  for (int i = 0; i < N_LAYERS; ++i) {
    tp_fill(tAm[i], AMt[i], AMt[i + 1]);
    tp_fill(tGm[i], AMt[i + 1], RMt[i + 1]);
    tp_fill(tRm[i], RMt[i + 1], RMt[i + 1]);
    gBond[i] = mk_fc(Lbond[i], H_EDGE, H_EDGE, tAm[i].numel);
    gRad[i]  = mk_fc(Lrad[i],  H_EDGE, H_EDGE, tAm[i].numel);
    gAgg[i]  = mk_fc(Lagg[i],  H_EDGE, H_EDGE, tGm[i].numel);
    gRes[i]  = mk_fc(Lres[i],  32,     32,     tRm[i].numel);
  }
  // embed MLPs (w1: din x dh, w2: dh x dh)
  const _Float16* rese_w1 = pack_w(p_rese_e.w1, 106, 32, 128, 32);
  const float*    rese_b1 = pack_bias(p_rese_e.b1, 32, 32);
  const _Float16* rese_w2 = pack_w(p_rese_e.w2, 32, 32, 32, 32);
  const float*    rese_b2 = pack_bias(p_rese_e.b2, 32, 32);
  const _Float16* res_w1  = pack_w(p_res_e.w1, 21, 32, 32, 32);
  const float*    res_b1  = pack_bias(p_res_e.b1, 32, 32);
  const _Float16* res_w2  = pack_w(p_res_e.w2, 32, 32, 32, 32);
  const float*    res_b2  = pack_bias(p_res_e.b2, 32, 32);
  const _Float16* agg_w1  = pack_w(p_agg_e.w1, 32, 16, 32, 32);
  const float*    agg_b1  = pack_bias(p_agg_e.b1, 16, 32);
  const _Float16* agg_w2  = pack_w(p_agg_e.w2, 16, 16, 32, 32);
  const float*    agg_b2  = pack_bias(p_agg_e.b2, 16, 32);
  const _Float16* rad_w1  = pack_w(p_rad_e.w1, 48, 16, 64, 32);
  const float*    rad_b1  = pack_bias(p_rad_e.b1, 16, 32);
  const _Float16* rad_w2  = pack_w(p_rad_e.w2, 16, 16, 32, 32);
  const float*    rad_b2  = pack_bias(p_rad_e.b2, 16, 32);
  const _Float16* bond_w1 = pack_w(p_bond_e.w1, 56, 16, 64, 32);
  const float*    bond_b1 = pack_bias(p_bond_e.b1, 16, 32);
  const _Float16* bond_w2 = pack_w(p_bond_e.w2, 16, 16, 32, 32);
  const float*    bond_b2 = pack_bias(p_bond_e.b2, 16, 32);
  const _Float16* atom_w1 = pack_w(p_atom_e.w1, 16, 16, 32, 32);
  const float*    atom_b1 = pack_bias(p_atom_e.b1, 16, 32);
  const _Float16* atom_w2 = pack_w(p_atom_e.w2, 16, 16, 32, 16);
  const float*    atom_b2 = pack_bias(p_atom_e.b2, 16, 16);

  // ---- features & embeddings ----
  k_onehot<<<gr(NR * 32), 256, 0, stream>>>(seq, res_in, NR, NUM_AA, 32);
  k_pad_rows<<<gr(NA * 32), 256, 0, stream>>>(atom_props, ap_pad, NA, 16, 32);

  k_res_edge_feat<<<gr(ERES), 256, 0, stream>>>(res_coords, sei, res_in,
                                                res_edge_in, res_sh, ERES);
  gemm(res_edge_in, rese_w1, rese_b1, h_buf, ERES, 128, 32, 1, stream);
  gemm(h_buf, rese_w2, rese_b2, res_edge_f, ERES, 32, 32, 0, stream);
  k_ln<<<gr(ERES), 256, 0, stream>>>(res_edge_f, p_rese_e.g, p_rese_e.b, ERES, 32, 32);

  gemm(res_in, res_w1, res_b1, h_buf, NR, 32, 32, 1, stream);
  gemm(h_buf, res_w2, res_b2, res_fA, NR, 32, 32, 0, stream);
  k_ln<<<gr(NR), 256, 0, stream>>>(res_fA, p_res_e.g, p_res_e.b, NR, 32, 32);

  k_agg_feat<<<gr(NA), 256, 0, stream>>>(atom_coords, res_coords, arb,
                                         atom_props, agg_in, agg_sh, NA);
  gemm(agg_in, agg_w1, agg_b1, h_buf, NA, 32, 32, 1, stream);
  gemm(h_buf, agg_w2, agg_b2, agg_edge_f, NA, 32, 32, 0, stream);
  k_ln<<<gr(NA), 256, 0, stream>>>(agg_edge_f, p_agg_e.g, p_agg_e.b, NA, 16, 32);

  k_rad_feat<<<gr(ER), 256, 0, stream>>>(atom_coords, rei, atom_props, rad_in,
                                         atom_sh + (size_t)EB * 4, ER);
  gemm(rad_in, rad_w1, rad_b1, h_buf, ER, 64, 32, 1, stream);
  gemm(h_buf, rad_w2, rad_b2, rad_edge_f, ER, 32, 32, 0, stream);
  k_ln<<<gr(ER), 256, 0, stream>>>(rad_edge_f, p_rad_e.g, p_rad_e.b, ER, 16, 32);

  k_bond_feat<<<gr(EB), 256, 0, stream>>>(atom_coords, bei, atom_props,
                                          bond_props, bond_in, atom_sh, EB);
  gemm(bond_in, bond_w1, bond_b1, h_buf, EB, 64, 32, 1, stream);
  gemm(h_buf, bond_w2, bond_b2, bond_edge_f, EB, 32, 32, 0, stream);
  k_ln<<<gr(EB), 256, 0, stream>>>(bond_edge_f, p_bond_e.g, p_bond_e.b, EB, 16, 32);

  gemm(ap_pad, atom_w1, atom_b1, h_buf, NA, 32, 32, 1, stream);
  gemm(h_buf, atom_w2, atom_b2, atom_fA, NA, 32, 16, 0, stream);
  k_ln<<<gr(NA), 256, 0, stream>>>(atom_fA, p_atom_e.g, p_atom_e.b, NA, 16, 16);

  // ---- layer loop ----
  float* atom_cur = atom_fA; float* atom_nxt = atom_fB;
  float* res_cur  = res_fA;  float* res_nxt  = res_fB;

  for (int i = 0; i < N_LAYERS; ++i) {
    // ===== atom conv over [bond | radius] edges =====
    TpMeta& tA = tAm[i];
    hipMemsetAsync(seg_sum_a, 0, (size_t)NA * tA.dout * sizeof(float), stream);
    hipMemsetAsync(seg_cnt_a, 0, (size_t)NA * sizeof(float), stream);
    int chunk = (int)(WCAP / tA.numel) & ~15; if (chunk > 8192) chunk = 8192;
    for (int cs = 0; cs < EB; cs += chunk) {
      int C = (EB - cs < chunk) ? (EB - cs) : chunk;
      gemm(bond_edge_f + (size_t)cs * 32, gBond[i].w1, gBond[i].b1, h_buf,
           C, 32, 32, 1, stream);
      gemm(h_buf, gBond[i].w2, gBond[i].b2, wbuf, C, 32, tA.numel, 0, stream);
      k_tp_scatter<<<gr(C), 256, 0, stream>>>(atom_cur, bei + EB + cs, 0,
          bei + cs, atom_sh + (size_t)cs * 4, wbuf, seg_sum_a, seg_cnt_a, C, tA);
    }
    for (int cs = 0; cs < ER; cs += chunk) {
      int C = (ER - cs < chunk) ? (ER - cs) : chunk;
      gemm(rad_edge_f + (size_t)cs * 32, gRad[i].w1, gRad[i].b1, h_buf,
           C, 32, 32, 1, stream);
      gemm(h_buf, gRad[i].w2, gRad[i].b2, wbuf, C, 32, tA.numel, 0, stream);
      k_tp_scatter<<<gr(C), 256, 0, stream>>>(atom_cur, rei + ER + cs, 0,
          rei + cs, atom_sh + (size_t)(EB + cs) * 4, wbuf, seg_sum_a, seg_cnt_a,
          C, tA);
    }
    k_seg_finalize<<<gr(NA * tA.dout), 256, 0, stream>>>(
        seg_sum_a, seg_cnt_a, atom_cur, atom_nxt, NA, tA.dout, tA.din);
    { float* t = atom_cur; atom_cur = atom_nxt; atom_nxt = t; }

    // ===== atom -> residue aggregation =====
    TpMeta& tG = tGm[i];
    hipMemsetAsync(seg_sum_r, 0, (size_t)NR * tG.dout * sizeof(float), stream);
    hipMemsetAsync(seg_cnt_r, 0, (size_t)NR * sizeof(float), stream);
    chunk = (int)(WCAP / tG.numel) & ~15; if (chunk > 8192) chunk = 8192;
    for (int cs = 0; cs < NA; cs += chunk) {
      int C = (NA - cs < chunk) ? (NA - cs) : chunk;
      gemm(agg_edge_f + (size_t)cs * 32, gAgg[i].w1, gAgg[i].b1, h_buf,
           C, 32, 32, 1, stream);
      gemm(h_buf, gAgg[i].w2, gAgg[i].b2, wbuf, C, 32, tG.numel, 0, stream);
      k_tp_scatter<<<gr(C), 256, 0, stream>>>(atom_cur, nullptr, cs,
          arb + cs, agg_sh + (size_t)cs * 4, wbuf, seg_sum_r, seg_cnt_r, C, tG);
    }
    k_seg_finalize<<<gr(NR * tG.dout), 256, 0, stream>>>(
        seg_sum_r, seg_cnt_r, res_cur, res_nxt, NR, tG.dout, irdim(RMt[i]));
    { float* t = res_cur; res_cur = res_nxt; res_nxt = t; }

    // ===== residue conv =====
    TpMeta& tR = tRm[i];
    hipMemsetAsync(seg_sum_r, 0, (size_t)NR * tR.dout * sizeof(float), stream);
    hipMemsetAsync(seg_cnt_r, 0, (size_t)NR * sizeof(float), stream);
    chunk = (int)(WCAP / tR.numel) & ~15; if (chunk > 8192) chunk = 8192;
    for (int cs = 0; cs < ERES; cs += chunk) {
      int C = (ERES - cs < chunk) ? (ERES - cs) : chunk;
      gemm(res_edge_f + (size_t)cs * 32, gRes[i].w1, gRes[i].b1, h_buf,
           C, 32, 32, 1, stream);
      gemm(h_buf, gRes[i].w2, gRes[i].b2, wbuf, C, 32, tR.numel, 0, stream);
      k_tp_scatter<<<gr(C), 256, 0, stream>>>(res_cur, sei + ERES + cs, 0,
          sei + cs, res_sh + (size_t)cs * 4, wbuf, seg_sum_r, seg_cnt_r, C, tR);
    }
    k_seg_finalize<<<gr(NR * tR.dout), 256, 0, stream>>>(
        seg_sum_r, seg_cnt_r, res_cur, res_nxt, NR, tR.dout, irdim(RMt[i + 1]));
    { float* t = res_cur; res_cur = res_nxt; res_nxt = t; }
  }

  size_t want = (size_t)out_size * sizeof(float);
  size_t have = (size_t)NR * 112 * sizeof(float);
  hipMemcpyAsync(d_out, res_cur, want < have ? want : have,
                 hipMemcpyDeviceToDevice, stream);
}
